// LSTM_9775345566003
// MI455X (gfx1250) — matrix operand order repaired and checked by reading, not compiled, not run on hardware
//
#include <hip/hip_runtime.h>
#include <hip/hip_bf16.h>
#include <stdint.h>

// Problem constants (from reference): T,B,I,H,L
#define T_ 512
#define B_ 32
#define I_ 1024
#define H_ 1024
#define L_ 2
#define NGATE 4
#define NWG (L_ * (H_ / 16))            // 128 persistent workgroups in recurrent phase

typedef __attribute__((ext_vector_type(16))) __bf16        v16bf;
typedef __attribute__((ext_vector_type(8)))  float         v8f;
typedef __attribute__((ext_vector_type(4))) unsigned int   u32x4;
typedef __attribute__((ext_vector_type(8))) unsigned int   u32x8;

__device__ __forceinline__ v16bf combine16(u32x4 a, u32x4 b) {
  u32x8 t;
  t[0] = a[0]; t[1] = a[1]; t[2] = a[2]; t[3] = a[3];
  t[4] = b[0]; t[5] = b[1]; t[6] = b[2]; t[7] = b[3];
  return __builtin_bit_cast(v16bf, t);
}

__device__ __forceinline__ v16bf as_v16bf(u32x8 t) {
  return __builtin_bit_cast(v16bf, t);
}

__device__ __forceinline__ unsigned short f2bf(float f) {
  unsigned int u = __float_as_uint(f);
  u += 0x7FFFu + ((u >> 16) & 1u);      // round-to-nearest-even
  return (unsigned short)(u >> 16);
}

__device__ __forceinline__ float sigmoidf_(float x) {
  return 1.0f / (1.0f + __expf(-x));
}

// ------------------------------------------------ Tensor Data Mover helpers -
// D# layout per CDNA5 ISA: group0 = {flags/count,
// lds_addr, global_addr, type=2}; group1 = {data_size, tensor dims/strides,
// tile dims}. data_size=8B elements. Generic LDS pointers carry the LDS byte
// offset in their low 32 bits (aperture rules, §10.2).
// This toolchain has the 6-arg builtin:
//   (uint32x4 g0, int32x8 g1, int32x4 g2, int32x4 g3, int32x8 extra, i32 cpol)
#if __has_builtin(__builtin_amdgcn_tensor_load_to_lds) && __has_builtin(__builtin_amdgcn_s_wait_tensorcnt)
#define HAVE_TDM 1
typedef __attribute__((ext_vector_type(4))) unsigned int tdm_v4u;
typedef __attribute__((ext_vector_type(8))) int          tdm_v8i;
typedef __attribute__((ext_vector_type(4))) int          tdm_v4i;

__device__ __forceinline__ void tdm_load_2d(const void* lds_ptr, const void* gptr,
                                            unsigned tile0_e8, unsigned tile1,
                                            unsigned long long stride0_e8) {
  unsigned long long ga = (unsigned long long)(uintptr_t)gptr;
  unsigned lds_off = (unsigned)(uintptr_t)lds_ptr;        // low 32 bits = LDS offset
  tdm_v4u g0;
  g0[0] = 1u;                                             // count=1, user descriptor
  g0[1] = lds_off;                                        // lds_addr (bytes)
  g0[2] = (unsigned)(ga & 0xFFFFFFFFull);                 // global_addr[31:0]
  g0[3] = (unsigned)((ga >> 32) & 0x01FFFFFFull) | 0x80000000u;  // addr[56:32] | type=2
  tdm_v8i g1;
  g1[0] = (int)(3u << 16);                                // data_size=8B, wg_mask=0
  g1[1] = (int)((tile0_e8 & 0xFFFFu) << 16);              // tensor_dim0[15:0]
  g1[2] = (int)(((tile0_e8 >> 16) & 0xFFFFu) | ((tile1 & 0xFFFFu) << 16)); // td0 hi | td1 lo
  g1[3] = (int)((tile0_e8 & 0xFFFFu) << 16);              // td1 hi(0) | tile_dim0
  g1[4] = (int)(tile1 & 0xFFFFu);                         // tile_dim1 | tile_dim2=0
  g1[5] = (int)(stride0_e8 & 0xFFFFFFFFull);              // dim0 stride lo32
  g1[6] = (int)((stride0_e8 >> 32) & 0xFFFFull);          // stride hi16 | dim1 stride lo(0)
  g1[7] = 0;
  tdm_v4i z4 = {0, 0, 0, 0};
  tdm_v8i z8 = {0, 0, 0, 0, 0, 0, 0, 0};
  __builtin_amdgcn_tensor_load_to_lds(g0, g1, z4, z4, z8, 0);
}
#else
#define HAVE_TDM 0
#endif

// ---------------------------------------------------------------- converts --
__global__ void __launch_bounds__(256)
f32_to_bf16_kernel(const float* __restrict__ src, unsigned short* __restrict__ dst, int n) {
  int i = blockIdx.x * 256 + threadIdx.x;
  if (i < n) dst[i] = f2bf(src[i]);
}

__global__ void init_counter_kernel(unsigned int* cnt) { *cnt = 0u; }

// --------------------------------------------------- phase 1: input GEMM ----
// xproj[l][t][g][b][h] = x[t,b,:] . W[g][l][h,:] + bias[g][l,h]
// C = [M=T*B=16384, N=4*H=4096] per layer, K=I=1024, bf16 WMMA, f32 accum.
__global__ void __launch_bounds__(256)
lstm_xproj_gemm(const unsigned short* __restrict__ xbf,
                const unsigned short* __restrict__ wbf,   // [g][l][hout][k]
                const float* __restrict__ bf_, const float* __restrict__ bi_,
                const float* __restrict__ bo_, const float* __restrict__ bc_,
                float* __restrict__ xproj)
{
  __shared__ __align__(32) unsigned short alds[64 * 32];   // A: 64 rows x 32 k
  __shared__ __align__(32) unsigned short blds[128 * 32];  // B: 128 cols x 32 k

  const int tid  = threadIdx.x;
  const int lane = tid & 31;
  const int w    = tid >> 5;
  const int m0   = blockIdx.x * 64;
  const int n0   = blockIdx.y * 128;
  const int l    = blockIdx.z;

  v8f acc[4];
#pragma unroll
  for (int i = 0; i < 4; ++i) acc[i] = {};

  for (int kk = 0; kk < I_ / 32; ++kk) {
    { // A panel: 64 rows x 64B, one 16B chunk per thread
      int row = tid >> 2, ch = tid & 3;
      *(u32x4*)&alds[row * 32 + ch * 8] =
          *(const u32x4*)&xbf[(size_t)(m0 + row) * I_ + kk * 32 + ch * 8];
    }
#pragma unroll
    for (int it = 0; it < 2; ++it) { // B panel: 128 cols x 64B
      int c = tid + it * 256;
      int nloc = c >> 2, ch = c & 3;
      int n = n0 + nloc;
      int g = n >> 10, hout = n & (H_ - 1);
      *(u32x4*)&blds[nloc * 32 + ch * 8] =
          *(const u32x4*)&wbf[(((size_t)g * L_ + l) * H_ + hout) * I_ + kk * 32 + ch * 8];
    }
    __syncthreads();
    {
      int n = w * 16 + (lane & 15);
      int koff = (lane >> 4) * 8;
      v16bf bt = combine16(*(const u32x4*)&blds[n * 32 + koff],
                           *(const u32x4*)&blds[n * 32 + 16 + koff]);
      v16bf at[4];
#pragma unroll
      for (int mt = 0; mt < 4; ++mt) {     // load all A tiles first (ILP)
        int m = mt * 16 + (lane & 15);
        int klo = (lane >> 4) * 8;
        at[mt] = combine16(*(const u32x4*)&alds[m * 32 + klo],
                           *(const u32x4*)&alds[m * 32 + 16 + klo]);
      }
#pragma unroll
      for (int mt = 0; mt < 4; ++mt)
        acc[mt] = __builtin_amdgcn_wmma_f32_16x16x32_bf16(
            false, at[mt], false, bt, (short)0, acc[mt], false, false);
    }
    __syncthreads();
  }

  // epilogue: + bias, scatter into xproj[l][t][g][b][h]
  const int N = n0 + w * 16 + (lane & 15);
  const int g = N >> 10;
  const int h = N & (H_ - 1);
  const float* bias = (g == 0) ? bf_ : (g == 1) ? bi_ : (g == 2) ? bo_ : bc_;
  const float bv = bias[l * H_ + h];
#pragma unroll
  for (int mt = 0; mt < 4; ++mt) {
#pragma unroll
    for (int r = 0; r < 8; ++r) {
      int m = m0 + mt * 16 + r + 8 * (lane >> 4);   // WMMA C layout
      int t = m >> 5, b = m & (B_ - 1);
      xproj[((((size_t)l * T_ + t) * NGATE + g) * B_ + b) * H_ + h] = acc[mt][r] + bv;
    }
  }
}

// ------------------------------------------------ phase 2: recurrence -------
// 128 persistent WGs: WG = (layer l, 16-wide hidden slice). U slice (4x16x1024
// bf16 = 128KB) is LDS-resident for the whole kernel (320KB/WGP on CDNA5),
// loaded via one TDM descriptor. Per step: TDM-stage h[l] (64KB bf16) to LDS,
// 8 waves = (4 gates x 2 m-tiles), each wave: 32 WMMAs over K=1024 spread over
// 4 interleaved accumulators (breaks the XDL RAW chain), then fused gate math
// with register-resident c state; device-wide atomic barrier between steps.
#define SMEM_U  (NGATE * 16 * H_ * 2)          // 131072 B
#define SMEM_H  (B_ * H_ * 2)                  //  65536 B
#define SMEM_G  (NGATE * B_ * 16 * 4)          //   8192 B
#define SMEM_TOTAL (SMEM_U + SMEM_H + SMEM_G)  // 204800 B

__global__ void __launch_bounds__(256)
lstm_recurrent(const unsigned short* __restrict__ ubf,    // [g][l][hout][hin]
               const float* __restrict__ xproj,
               const float* __restrict__ c0,
               unsigned short* __restrict__ hbuf0,
               unsigned short* __restrict__ hbuf1,
               unsigned int* __restrict__ cnt,
               float* __restrict__ out)
{
  extern __shared__ __attribute__((aligned(128))) unsigned char smem[];
  unsigned short* u_lds = (unsigned short*)smem;
  unsigned short* h_lds = (unsigned short*)(smem + SMEM_U);
  float*          gbuf  = (float*)(smem + SMEM_U + SMEM_H);

  const int tid   = threadIdx.x;
  const int lane  = tid & 31;
  const int wv    = tid >> 5;
  const int gg    = wv >> 1;                 // gate for this wave
  const int mt    = wv & 1;                  // batch m-tile (rows 16*mt..)
  const int l     = blockIdx.x >> 6;
  const int slice = blockIdx.x & 63;
  const int h0    = slice * 16;

  // ---- Preload U slice into LDS once: 4 gates x 16 rows x 1024 k (bf16).
  // Each gate's 16-row chunk is 32KB contiguous; gate blocks are strided:
  // one 2D TDM tile (tile0=32KB/8, tile1=4 gates).
#if HAVE_TDM
  if (wv == 0) {
    tdm_load_2d(u_lds, ubf + ((size_t)l * H_ + h0) * H_,
                /*tile0_e8=*/16 * H_ * 2 / 8, /*tile1=*/NGATE,
                /*stride0_e8=*/(unsigned long long)L_ * H_ * H_ * 2 / 8);
    __builtin_amdgcn_s_wait_tensorcnt(0);
  }
#else
  for (int idx = tid; idx < NGATE * 16 * (H_ / 8); idx += 256) {
    int row = idx >> 7;
    int ch  = idx & 127;
    int g = row >> 4, n = row & 15;
    *(u32x4*)&u_lds[row * H_ + ch * 8] =
        *(const u32x4*)&ubf[(((size_t)g * L_ + l) * H_ + (h0 + n)) * H_ + ch * 8];
  }
#endif

  // Register-resident cell state: thread owns elements (b,hl) and (b,hl+1)
  const int e0  = tid * 2;
  const int eb  = e0 >> 4;
  const int ehl = e0 & 15;
  float cst[2];
#pragma unroll
  for (int q = 0; q < 2; ++q)
    cst[q] = c0[((size_t)l * B_ + eb) * H_ + h0 + ehl + q];

  unsigned short* hb[2] = { hbuf0, hbuf1 };
  float* out_h = out + (size_t)T_ * B_ * H_;
  float* out_c = out_h + (size_t)L_ * B_ * H_;

  __syncthreads();

  for (int t = 0; t < T_; ++t) {
    const unsigned short* hcur = hb[t & 1];
    unsigned short*       hnxt = hb[(t + 1) & 1];

    // ---- Stage h[l] (32 x 1024 bf16 = 64KB, fully contiguous) into LDS
#if HAVE_TDM
    if (wv == 0) {
      tdm_load_2d(h_lds, hcur + (size_t)l * B_ * H_,
                  /*tile0_e8=*/B_ * H_ * 2 / 8, /*tile1=*/1,
                  /*stride0_e8=*/(unsigned long long)B_ * H_ * 2 / 8);
      __builtin_amdgcn_s_wait_tensorcnt(0);
    }
#else
    for (int idx = tid; idx < B_ * (H_ / 8); idx += 256) {
      int row = idx >> 7, ch = idx & 127;
      *(u32x4*)&h_lds[row * H_ + ch * 8] =
          *(const u32x4*)&hcur[((size_t)l * B_ + row) * H_ + ch * 8];
    }
#endif
    __syncthreads();

    // ---- K=1024 as 32 WMMAs over 4 interleaved accumulators
    v8f acc4[4];
#pragma unroll
    for (int i = 0; i < 4; ++i) acc4[i] = {};
#pragma unroll 4
    for (int kk = 0; kk < H_ / 32; ++kk) {
      int n    = lane & 15;
      int koff = kk * 32 + (lane >> 4) * 8;
      v16bf bt = combine16(*(const u32x4*)&u_lds[(gg * 16 + n) * H_ + koff],
                           *(const u32x4*)&u_lds[(gg * 16 + n) * H_ + koff + 16]);
      int m    = mt * 16 + (lane & 15);
      int klo  = kk * 32 + (lane >> 4) * 8;
      v16bf at = combine16(*(const u32x4*)&h_lds[m * H_ + klo],
                           *(const u32x4*)&h_lds[m * H_ + klo + 16]);
      acc4[kk & 3] = __builtin_amdgcn_wmma_f32_16x16x32_bf16(
          false, at, false, bt, (short)0, acc4[kk & 3], false, false);
    }

    // Stage gate tile to LDS in (gate, batch, hl) order
#pragma unroll
    for (int r = 0; r < 8; ++r) {
      int b = mt * 16 + r + 8 * (lane >> 4);
      gbuf[(gg * B_ + b) * 16 + (lane & 15)] =
          (acc4[0][r] + acc4[1][r]) + (acc4[2][r] + acc4[3][r]);
    }
    __syncthreads();

    // ---- Fused elementwise LSTM cell (each thread: 2 adjacent elements)
    const size_t xbase = (((size_t)l * T_ + t) * NGATE) * B_ * H_;
    const size_t xi0   = xbase + (size_t)eb * H_ + h0 + ehl;
    float2 xg[NGATE], gg2[NGATE];
#pragma unroll
    for (int g = 0; g < NGATE; ++g) {
      xg[g]  = *(const float2*)&xproj[xi0 + (size_t)g * B_ * H_];
      gg2[g] = *(const float2*)&gbuf[(g * B_ + eb) * 16 + ehl];
    }
    if (t + 1 < T_) {                 // prefetch next step's xproj slice
      const float* nx = xproj + xi0 + (size_t)NGATE * B_ * H_;
      __builtin_prefetch(nx, 0, 1);
      __builtin_prefetch(nx + (size_t)B_ * H_, 0, 1);
      __builtin_prefetch(nx + 2 * (size_t)B_ * H_, 0, 1);
      __builtin_prefetch(nx + 3 * (size_t)B_ * H_, 0, 1);
    }
#pragma unroll
    for (int q = 0; q < 2; ++q) {
      int b = eb, hl = ehl + q;
      float gf = (q ? xg[0].y : xg[0].x) + (q ? gg2[0].y : gg2[0].x);
      float gi = (q ? xg[1].y : xg[1].x) + (q ? gg2[1].y : gg2[1].x);
      float go = (q ? xg[2].y : xg[2].x) + (q ? gg2[2].y : gg2[2].x);
      float gc = (q ? xg[3].y : xg[3].x) + (q ? gg2[3].y : gg2[3].x);
      float f  = sigmoidf_(gf);
      float i_ = sigmoidf_(gi);
      float o  = sigmoidf_(go);
      float ct = tanhf(gc);
      float cn = f * cst[q] + i_ * ct;
      cst[q] = cn;
      float hv = o * tanhf(cn);
      hnxt[((size_t)l * B_ + b) * H_ + h0 + hl] = f2bf(hv);
      if (l == L_ - 1)
        out[((size_t)t * B_ + b) * H_ + h0 + hl] = hv;
      if (t == T_ - 1) {
        out_h[((size_t)l * B_ + b) * H_ + h0 + hl] = hv;
        out_c[((size_t)l * B_ + b) * H_ + h0 + hl] = cn;
      }
    }

    // ---- Device-wide barrier between timesteps (counter reset each launch)
    __syncthreads();
    if (tid == 0) {
      __threadfence();
      atomicAdd(cnt, 1u);
      unsigned target = (unsigned)(t + 1) * NWG;
      while (__hip_atomic_load(cnt, __ATOMIC_RELAXED, __HIP_MEMORY_SCOPE_AGENT) < target) {
        __builtin_amdgcn_s_sleep(1);
      }
      __threadfence();
    }
    __syncthreads();
  }
}

// ----------------------------------------------------------- launcher -------
extern "C" void kernel_launch(void* const* d_in, const int* in_sizes, int n_in,
                              void* d_out, int out_size, void* d_ws, size_t ws_size,
                              hipStream_t stream) {
  (void)in_sizes; (void)n_in; (void)out_size; (void)ws_size;

  const float* x  = (const float*)d_in[0];
  const float* h0 = (const float*)d_in[1];
  const float* c0 = (const float*)d_in[2];
  const float* W[4]    = { (const float*)d_in[3],  (const float*)d_in[4],
                           (const float*)d_in[5],  (const float*)d_in[6] };
  const float* U[4]    = { (const float*)d_in[7],  (const float*)d_in[8],
                           (const float*)d_in[9],  (const float*)d_in[10] };
  const float* bias[4] = { (const float*)d_in[11], (const float*)d_in[12],
                           (const float*)d_in[13], (const float*)d_in[14] };
  float* out = (float*)d_out;

  // Workspace carve-up
  char* ws = (char*)d_ws;
  size_t off = 0;
  float* xproj = (float*)(ws + off);          off += (size_t)L_ * T_ * NGATE * B_ * H_ * sizeof(float);
  unsigned short* xbf   = (unsigned short*)(ws + off); off += (size_t)T_ * B_ * I_ * 2;
  unsigned short* wbf   = (unsigned short*)(ws + off); off += (size_t)NGATE * L_ * H_ * I_ * 2;
  unsigned short* ubf   = (unsigned short*)(ws + off); off += (size_t)NGATE * L_ * H_ * H_ * 2;
  unsigned short* hbuf0 = (unsigned short*)(ws + off); off += (size_t)L_ * B_ * H_ * 2;
  unsigned short* hbuf1 = (unsigned short*)(ws + off); off += (size_t)L_ * B_ * H_ * 2;
  unsigned int*   cnt   = (unsigned int*)(ws + off);   off += 64;

  // bf16 conversions (one-time per launch)
  {
    int n = T_ * B_ * I_;
    f32_to_bf16_kernel<<<(n + 255) / 256, 256, 0, stream>>>(x, xbf, n);
  }
  for (int g = 0; g < 4; ++g) {
    int n = L_ * H_ * I_;
    f32_to_bf16_kernel<<<(n + 255) / 256, 256, 0, stream>>>(W[g], wbf + (size_t)g * n, n);
  }
  for (int g = 0; g < 4; ++g) {
    int n = L_ * H_ * H_;
    f32_to_bf16_kernel<<<(n + 255) / 256, 256, 0, stream>>>(U[g], ubf + (size_t)g * n, n);
  }
  {
    int n = L_ * B_ * H_;
    f32_to_bf16_kernel<<<(n + 255) / 256, 256, 0, stream>>>(h0, hbuf0, n);
  }
  init_counter_kernel<<<1, 1, 0, stream>>>(cnt);

  // Phase 1: all input projections as one batched GEMM
  dim3 g1((T_ * B_) / 64, (NGATE * H_) / 128, L_);
  lstm_xproj_gemm<<<g1, 256, 0, stream>>>(xbf, wbf, bias[0], bias[1], bias[2], bias[3], xproj);

  // Phase 2: persistent recurrent kernel with LDS-resident U slices
  (void)hipFuncSetAttribute((const void*)lstm_recurrent,
                            hipFuncAttributeMaxDynamicSharedMemorySize, SMEM_TOTAL);
  lstm_recurrent<<<NWG, 256, SMEM_TOTAL, stream>>>(ubf, xproj, c0, hbuf0, hbuf1, cnt, out);
}
